// QuantizedLinear_86371792322993
// MI455X (gfx1250) — compile-verified, matrix operand order checked by reading, not run on hardware
//
#include <hip/hip_runtime.h>

typedef __attribute__((ext_vector_type(16))) _Float16 v16h;
typedef __attribute__((ext_vector_type(8)))  _Float16 v8h;
typedef __attribute__((ext_vector_type(8)))  float    v8f;

#define M_TOTAL 256          // 8 * 32 rows of x
#define K_TOTAL 4096
#define N_TOTAL 12288
#define NG      32           // number of 128-wide scale groups
#define KITER   64           // K per staged slice (2 WMMA sub-steps)
#define NITERS  (K_TOTAL / KITER)   // 64
#define LDS_ROW 72           // 64 f16 of data + 8 f16 pad (bank spread)

// Async copy of 128 contiguous bytes global -> LDS (8 x B128).
// the immediate applies to BOTH addresses. Tracked by ASYNCcnt.
__device__ __forceinline__ void async_copy_row128(unsigned lds_off, const void* gaddr) {
    asm volatile("global_load_async_to_lds_b128 %0, %1, off offset:0"
                 :: "v"(lds_off), "v"(gaddr) : "memory");
    asm volatile("global_load_async_to_lds_b128 %0, %1, off offset:16"
                 :: "v"(lds_off), "v"(gaddr) : "memory");
    asm volatile("global_load_async_to_lds_b128 %0, %1, off offset:32"
                 :: "v"(lds_off), "v"(gaddr) : "memory");
    asm volatile("global_load_async_to_lds_b128 %0, %1, off offset:48"
                 :: "v"(lds_off), "v"(gaddr) : "memory");
    asm volatile("global_load_async_to_lds_b128 %0, %1, off offset:64"
                 :: "v"(lds_off), "v"(gaddr) : "memory");
    asm volatile("global_load_async_to_lds_b128 %0, %1, off offset:80"
                 :: "v"(lds_off), "v"(gaddr) : "memory");
    asm volatile("global_load_async_to_lds_b128 %0, %1, off offset:96"
                 :: "v"(lds_off), "v"(gaddr) : "memory");
    asm volatile("global_load_async_to_lds_b128 %0, %1, off offset:112"
                 :: "v"(lds_off), "v"(gaddr) : "memory");
}

__device__ __forceinline__ void wait_asynccnt0() {
    asm volatile("s_wait_asynccnt 0x0" ::: "memory");
}

// int8x16 (two 8-byte packs) -> f16x16, scaled; vector*scalar -> v_pk_mul_f16
__device__ __forceinline__ v16h dequant16(long q0, long q1, _Float16 s) {
    v16h q;
#pragma unroll
    for (int j = 0; j < 8; ++j)
        q[j] = (_Float16)(signed char)(q0 >> (8 * j));
#pragma unroll
    for (int j = 0; j < 8; ++j)
        q[8 + j] = (_Float16)(signed char)(q1 >> (8 * j));
    return q * s;
}

__device__ __forceinline__ v16h load_afrag(const _Float16* base) {
    v8h alo = *(const v8h*)base;          // K base + {0..7}
    v8h ahi = *(const v8h*)(base + 16);   // K base + 16 + {0..7}
    return __builtin_shufflevector(alo, ahi,
        0, 1, 2, 3, 4, 5, 6, 7, 8, 9, 10, 11, 12, 13, 14, 15);
}

__global__ __launch_bounds__(256) void qlinear_wmma_kernel(
    const _Float16*    __restrict__ x,       // [256, 4096] f16
    const _Float16*    __restrict__ scales,  // [12288, 32] f16
    const _Float16*    __restrict__ bias,    // [12288]     f16
    const signed char* __restrict__ wq,      // [12288, 4096] int8
    _Float16*          __restrict__ out)     // [256, 12288] f16
{
    // double-buffered x slice: 256 rows x 64 f16 (+pad) per buffer (~72 KB)
    __shared__ _Float16 xs[2][M_TOTAL * LDS_ROW];

    const int tid  = threadIdx.x;
    const int wave = tid >> 5;
    const int lane = tid & 31;
    const int col  = lane & 15;   // N within tile / M within A tile
    const int half = lane >> 4;   // lane-half selects K sub-range

    // this wave's 16 output features
    const int    n0   = blockIdx.x * 128 + wave * 16;
    const size_t wrow = (size_t)(n0 + col) * (size_t)K_TOTAL;

    v8f acc[16];
    const v8f vzero = {};
#pragma unroll
    for (int t = 0; t < 16; ++t) acc[t] = vzero;

    // each thread async-stages one x row's 128-byte K-slice per iteration
    const _Float16* xrowp = x + (size_t)tid * (size_t)K_TOTAL;
    const unsigned lds_off0 = (unsigned)(unsigned long long)&xs[0][tid * LDS_ROW];
    const unsigned lds_off1 = (unsigned)(unsigned long long)&xs[1][tid * LDS_ROW];

    // prologue: stage slice 0 into buffer 0
    async_copy_row128(lds_off0, xrowp);
    wait_asynccnt0();
    __syncthreads();

    for (int it = 0; it < NITERS; ++it) {
        const int buf = it & 1;

        // ---- async-stage next x slice into the other buffer (overlaps WMMA) ----
        if (it + 1 < NITERS) {
            async_copy_row128(buf ? lds_off0 : lds_off1,
                              xrowp + (size_t)(it + 1) * KITER);
        }

        // ---- load weights for both K sub-steps (4 x b64 per lane) ----
        // lane holds one N column; K interleave per 16-bit fragment layout:
        // half 0 -> K 0..7 & 16..23 ; half 1 -> K 8..15 & 24..31
        const signed char* wp = wq + wrow + (size_t)it * KITER;
        long q00 = *(const long*)(wp +      half * 8);
        long q01 = *(const long*)(wp + 16 + half * 8);
        long q10 = *(const long*)(wp + 32 + half * 8);
        long q11 = *(const long*)(wp + 48 + half * 8);
        __builtin_prefetch(wp + 2 * KITER, 0, 0);   // stream hint for weights

        // scale group: K=64 chunk it covers k in [64*it, 64*it+64) -> group it/2
        const _Float16 s = scales[(size_t)(n0 + col) * NG + (it >> 1)];

#pragma unroll
        for (int j = 0; j < 2; ++j) {
            const v16h bfrag = dequant16(j ? q10 : q00, j ? q11 : q01, s);

            // rotating A-fragment prefetch: keep next ds_load pair in flight
            const _Float16* abase = &xs[buf][col * LDS_ROW + j * 32 + half * 8];
            v16h a_cur = load_afrag(abase);
#pragma unroll
            for (int t = 0; t < 16; ++t) {
                v16h a_nxt = a_cur;
                if (t < 15)
                    a_nxt = load_afrag(abase + (t + 1) * 16 * LDS_ROW);
                acc[t] = __builtin_amdgcn_wmma_f32_16x16x32_f16(
                    /*neg_a=*/false, a_cur, /*neg_b=*/false, bfrag,
                    /*c_mod=*/(short)0, acc[t], /*reuse_a=*/false, /*reuse_b=*/false);
                a_cur = a_nxt;
            }
        }

        // ---- fence: own async copies done, then block-wide visibility ----
        wait_asynccnt0();
        __syncthreads();
    }

    // ---- epilogue: bias + convert + store ----
    // C layout: lanes 0-15 -> M = vgpr_i, lanes 16-31 -> M = 8 + vgpr_i; N = col
    const float bv = (float)bias[n0 + col];
#pragma unroll
    for (int t = 0; t < 16; ++t) {
#pragma unroll
        for (int i = 0; i < 8; ++i) {
            const int m = t * 16 + half * 8 + i;
            out[(size_t)m * N_TOTAL + (n0 + col)] = (_Float16)(acc[t][i] + bv);
        }
    }
}

extern "C" void kernel_launch(void* const* d_in, const int* in_sizes, int n_in,
                              void* d_out, int out_size, void* d_ws, size_t ws_size,
                              hipStream_t stream) {
    // setup_inputs order: x (f16), scales (f16), bias (f16), weight_q (int8)
    const _Float16*    x      = (const _Float16*)d_in[0];
    const _Float16*    scales = (const _Float16*)d_in[1];
    const _Float16*    bias   = (const _Float16*)d_in[2];
    const signed char* wq     = (const signed char*)d_in[3];
    _Float16*          out    = (_Float16*)d_out;

    // 96 blocks x 8 waves x 16 cols = 12288 output features, each wave does all M=256
    qlinear_wmma_kernel<<<dim3(N_TOTAL / 128), dim3(256), 0, stream>>>(
        x, scales, bias, wq, out);
}
